// ContrastiveLoss_72516227826057
// MI455X (gfx1250) — matrix operand order repaired and checked by reading, not compiled, not run on hardware
//
#include <hip/hip_runtime.h>
#include <hip/hip_bf16.h>
#include <stdint.h>

typedef __attribute__((ext_vector_type(16))) __bf16 v16bf;
typedef __attribute__((ext_vector_type(8)))  __bf16 v8bf;
typedef __attribute__((ext_vector_type(4)))  __bf16 v4bf;
typedef __attribute__((ext_vector_type(8)))  float  v8f;
typedef __attribute__((ext_vector_type(4)))  float  f32x4;

#define N_ROWS 8192
#define DIM    256
#define BM     128          // rows per block (8 waves x 16)
#define BN     64           // cols per tile iteration
#define CSPLIT 16           // column splits across grid.y
#define COLS_PER_BLOCK (N_ROWS / CSPLIT)   // 512
#define NITER  (COLS_PER_BLOCK / BN)       // 8
#define TAU_INV 10.0f

// LDS layout per buffer (bytes): bhi[16 kk][64 row][16 e] = 32768 | blo = 32768
#define BUF_BYTES 65536u
#define ARR_BYTES 32768u

__device__ __forceinline__ void cvt_split(float f, __bf16& h, __bf16& l) {
    h = (__bf16)f;               // RNE to bf16 (high part)
    l = (__bf16)(f - (float)h);  // residual (low part)
}

// ---------------- kernel 0: zero the denominator accumulator ----------------
__global__ void cl_zero_kernel(float* __restrict__ d) {
    int i = blockIdx.x * blockDim.x + threadIdx.x;
    if (i < N_ROWS) d[i] = 0.0f;
}

// ---------------- kernel 1: inverse row norms (one wave32 per row) ----------
__global__ __launch_bounds__(256) void cl_norm_kernel(const float* __restrict__ x,
                                                      float* __restrict__ invn) {
    const int wave = threadIdx.x >> 5;
    const int lane = threadIdx.x & 31;
    const int row  = blockIdx.x * 8 + wave;
    const float* p = x + (size_t)row * DIM + lane * 8;
    f32x4 a = *(const f32x4*)p;
    f32x4 b = *(const f32x4*)(p + 4);
    float s = a.x*a.x + a.y*a.y + a.z*a.z + a.w*a.w
            + b.x*b.x + b.y*b.y + b.z*b.z + b.w*b.w;
#pragma unroll
    for (int m = 1; m < 32; m <<= 1) s += __shfl_xor(s, m, 32);
    if (lane == 0) invn[row] = rsqrtf(s);
}

// ---------------- kernel 1b: one-time fp32 -> split-bf16 conversion ---------
__global__ __launch_bounds__(256) void cl_convert_kernel(const float* __restrict__ x,
                                                         __bf16* __restrict__ xhi,
                                                         __bf16* __restrict__ xlo) {
    const int i = blockIdx.x * blockDim.x + threadIdx.x;   // float4 index
    f32x4 f = *(const f32x4*)(x + (size_t)i * 4);
    v4bf hv, lv;
    __bf16 h, l;
    cvt_split(f.x, h, l); hv[0] = h; lv[0] = l;
    cvt_split(f.y, h, l); hv[1] = h; lv[1] = l;
    cvt_split(f.z, h, l); hv[2] = h; lv[2] = l;
    cvt_split(f.w, h, l); hv[3] = h; lv[3] = l;
    *(v4bf*)(xhi + (size_t)i * 4) = hv;
    *(v4bf*)(xlo + (size_t)i * 4) = lv;
}

// Issue this thread's share (16 lanes x 16B) of one 64KB B-buffer fill via
// async global->LDS DMA. Chunk q in [0,4096): arr = q>>11 (hi/lo),
// w = q&2047 -> (kk = w>>7, r = (w&127)>>1, e0 = (w&1)*8). LDS fragment
// layout: element k of column-row r at ((k>>4)*64 + r)*16 + (k&15).
__device__ __forceinline__ void stage_async(const __bf16* __restrict__ xhi,
                                            const __bf16* __restrict__ xlo,
                                            int colblk, unsigned ldsbase, int tid) {
#pragma unroll
    for (int ci = 0; ci < 16; ++ci) {
        const int q   = ci * 256 + tid;        // 0..4095
        const int arr = q >> 11;               // 0 = hi, 1 = lo
        const int w   = q & 2047;              // 16B chunk within array
        const int kk  = w >> 7;
        const int r   = (w & 127) >> 1;
        const int e0  = (w & 1) * 8;
        const unsigned lds_off = ldsbase + (unsigned)arr * ARR_BYTES + (unsigned)w * 16u;
        const __bf16* src = (arr ? xlo : xhi)
                          + (size_t)(colblk + r) * DIM + kk * 16 + e0;
        asm volatile("global_load_async_to_lds_b128 %0, %1, off"
                     :: "v"(lds_off),
                        "v"((unsigned long long)(uintptr_t)src)
                     : "memory");
    }
}

__device__ __forceinline__ void wait_async0() {
    asm volatile("s_wait_asynccnt 0x0" ::: "memory");
}

// ---------------- kernel 2: fused Gram/softmax-denominator (async) ----------
// grid = (64 row-blocks, 16 col-splits), block = 256 threads (8 waves)
__global__ __launch_bounds__(256) void cl_main_async(const __bf16* __restrict__ xhi,
                                                     const __bf16* __restrict__ xlo,
                                                     const float* __restrict__ invn,
                                                     float* __restrict__ d,
                                                     float* __restrict__ spair) {
    extern __shared__ __bf16 smem[];   // 2 x 64KB double buffer

    const int tid   = threadIdx.x;
    const int wave  = tid >> 5;
    const int lane  = tid & 31;
    const int m16   = lane & 15;
    const int hi16  = lane >> 4;

    const int row0      = blockIdx.x * BM;
    const int colbase   = blockIdx.y * COLS_PER_BLOCK;
    const int grow_base = row0 + wave * 16;

    // ---- preload A fragments (bf16, already split) ----
    // A frag element e (chunk c): k = 32c + 16*hi16 + e
    v16bf a_hi[8], a_lo[8];
    {
        const int arow = grow_base + m16;
        const __bf16* hbase = xhi + (size_t)arow * DIM;
        const __bf16* lbase = xlo + (size_t)arow * DIM;
        union U { v16bf v; v8bf h[2]; };
#pragma unroll
        for (int c = 0; c < 8; ++c) {
            const int k0 = 32 * c + 16 * hi16;
            U uh, ul;
            uh.h[0] = *(const v8bf*)(hbase + k0);
            uh.h[1] = *(const v8bf*)(hbase + k0 + 8);
            ul.h[0] = *(const v8bf*)(lbase + k0);
            ul.h[1] = *(const v8bf*)(lbase + k0 + 8);
            a_hi[c] = uh.v;
            a_lo[c] = ul.v;
        }
    }

    float rowsum[8];
    float rn[8];
#pragma unroll
    for (int r = 0; r < 8; ++r) {
        rowsum[r] = 0.0f;
        rn[r]     = invn[grow_base + 8 * hi16 + r];
    }

    // prologue: stage iteration 0 into buffer 0
    stage_async(xhi, xlo, colbase, 0u, tid);
    wait_async0();
    __syncthreads();

    for (int it = 0; it < NITER; ++it) {
        const int cur    = it & 1;
        const int colblk = colbase + it * BN;

        // overlap: kick off next tile's DMA into the other buffer
        if (it + 1 < NITER)
            stage_async(xhi, xlo, colbase + (it + 1) * BN,
                        (unsigned)(1 - cur) * BUF_BYTES, tid);

        const __bf16* bhi = smem + (size_t)cur * (BUF_BYTES / 2);   // elements
        const __bf16* blo = bhi + (ARR_BYTES / 2);

        // ---- WMMA: 4 C-tiles of 16x16, K = 256 in chunks of 32 ----
        v8f acc[4] = {v8f{}, v8f{}, v8f{}, v8f{}};
#pragma unroll
        for (int c = 0; c < 8; ++c) {
#pragma unroll
            for (int t = 0; t < 4; ++t) {
                const int off = (((2 * c + hi16) * 64) + t * 16 + m16) * 16;
                const v16bf b_h = *(const v16bf*)(bhi + off);
                const v16bf b_l = *(const v16bf*)(blo + off);
                // split-bf16: Ah*Bh + Ah*Bl + Al*Bh
                acc[t] = __builtin_amdgcn_wmma_f32_16x16x32_bf16(
                    false, a_hi[c], false, b_h, (short)0, acc[t], false, false);
                acc[t] = __builtin_amdgcn_wmma_f32_16x16x32_bf16(
                    false, a_hi[c], false, b_l, (short)0, acc[t], false, false);
                acc[t] = __builtin_amdgcn_wmma_f32_16x16x32_bf16(
                    false, a_lo[c], false, b_h, (short)0, acc[t], false, false);
            }
        }

        // ---- epilogue: scale, mask diagonal, exp, pair capture ----
#pragma unroll
        for (int t = 0; t < 4; ++t) {
            const int gcol = colblk + t * 16 + m16;
            const float cin = invn[gcol];
#pragma unroll
            for (int r = 0; r < 8; ++r) {
                const int grow = grow_base + 8 * hi16 + r;
                const float s = acc[t][r] * rn[r] * cin;
                if (grow == gcol) {
                    rowsum[r] += 1.0f;            // masked diagonal: exp(0)
                } else {
                    rowsum[r] += __expf(s * TAU_INV);
                    if (((grow & 1) == 0) && (gcol == grow + 1)) {
                        spair[grow >> 1] = s;     // positive-pair logit
                    }
                }
            }
        }

        if (it + 1 < NITER) {
            wait_async0();       // my share of next buffer landed in LDS
            __syncthreads();     // everyone's share landed; prev reads done
        }
    }

    // ---- reduce row partials across the 16 lanes of each half-wave ----
#pragma unroll
    for (int r = 0; r < 8; ++r) {
#pragma unroll
        for (int m = 1; m < 16; m <<= 1)
            rowsum[r] += __shfl_xor(rowsum[r], m, 32);
    }
    if (m16 == 0) {
#pragma unroll
        for (int r = 0; r < 8; ++r)
            atomicAdd(&d[grow_base + 8 * hi16 + r], rowsum[r]);
    }
}

// ---------------- fallback kernel 2 (in-kernel conversion, small ws) --------
__global__ __launch_bounds__(256) void cl_main_sync(const float* __restrict__ x,
                                                    const float* __restrict__ invn,
                                                    float* __restrict__ d,
                                                    float* __restrict__ spair) {
    extern __shared__ __bf16 smem[];
    __bf16* bhi = smem;
    __bf16* blo = smem + 16 * 64 * 16;

    const int tid   = threadIdx.x;
    const int wave  = tid >> 5;
    const int lane  = tid & 31;
    const int m16   = lane & 15;
    const int hi16  = lane >> 4;

    const int row0      = blockIdx.x * BM;
    const int colbase   = blockIdx.y * COLS_PER_BLOCK;
    const int grow_base = row0 + wave * 16;

    v16bf a_hi[8], a_lo[8];
    {
        const int arow = grow_base + m16;
        const float* abase = x + (size_t)arow * DIM;
#pragma unroll
        for (int c = 0; c < 8; ++c) {
            const float* p0 = abase + 32 * c + 16 * hi16;
            const float* p1 = p0 + 8;
            f32x4 r0 = *(const f32x4*)p0;
            f32x4 r1 = *(const f32x4*)(p0 + 4);
            f32x4 r2 = *(const f32x4*)p1;
            f32x4 r3 = *(const f32x4*)(p1 + 4);
            float v[16] = { r0.x, r0.y, r0.z, r0.w, r1.x, r1.y, r1.z, r1.w,
                            r2.x, r2.y, r2.z, r2.w, r3.x, r3.y, r3.z, r3.w };
#pragma unroll
            for (int e = 0; e < 16; ++e) {
                __bf16 h, l;
                cvt_split(v[e], h, l);
                a_hi[c][e] = h;
                a_lo[c][e] = l;
            }
        }
    }

    float rowsum[8];
    float rn[8];
#pragma unroll
    for (int r = 0; r < 8; ++r) {
        rowsum[r] = 0.0f;
        rn[r]     = invn[grow_base + 8 * hi16 + r];
    }

    for (int it = 0; it < NITER; ++it) {
        const int colblk = colbase + it * BN;
        __syncthreads();
#pragma unroll
        for (int j = 0; j < 16; ++j) {
            const int idx = j * 256 + tid;
            const int r   = idx >> 6;
            const int kq  = idx & 63;
            f32x4 f = *(const f32x4*)(x + (size_t)(colblk + r) * DIM + kq * 4);
            const int k0  = kq * 4;
            const int off = (((k0 >> 4) * 64) + r) * 16 + (k0 & 15);
            v4bf hv, lv;
            __bf16 h, l;
            cvt_split(f.x, h, l); hv[0] = h; lv[0] = l;
            cvt_split(f.y, h, l); hv[1] = h; lv[1] = l;
            cvt_split(f.z, h, l); hv[2] = h; lv[2] = l;
            cvt_split(f.w, h, l); hv[3] = h; lv[3] = l;
            *(v4bf*)(bhi + off) = hv;
            *(v4bf*)(blo + off) = lv;
        }
        __syncthreads();

        v8f acc[4] = {v8f{}, v8f{}, v8f{}, v8f{}};
#pragma unroll
        for (int c = 0; c < 8; ++c) {
#pragma unroll
            for (int t = 0; t < 4; ++t) {
                const int off = (((2 * c + hi16) * 64) + t * 16 + m16) * 16;
                const v16bf b_h = *(const v16bf*)(bhi + off);
                const v16bf b_l = *(const v16bf*)(blo + off);
                acc[t] = __builtin_amdgcn_wmma_f32_16x16x32_bf16(
                    false, a_hi[c], false, b_h, (short)0, acc[t], false, false);
                acc[t] = __builtin_amdgcn_wmma_f32_16x16x32_bf16(
                    false, a_hi[c], false, b_l, (short)0, acc[t], false, false);
                acc[t] = __builtin_amdgcn_wmma_f32_16x16x32_bf16(
                    false, a_lo[c], false, b_h, (short)0, acc[t], false, false);
            }
        }

#pragma unroll
        for (int t = 0; t < 4; ++t) {
            const int gcol = colblk + t * 16 + m16;
            const float cin = invn[gcol];
#pragma unroll
            for (int r = 0; r < 8; ++r) {
                const int grow = grow_base + 8 * hi16 + r;
                const float s = acc[t][r] * rn[r] * cin;
                if (grow == gcol) {
                    rowsum[r] += 1.0f;
                } else {
                    rowsum[r] += __expf(s * TAU_INV);
                    if (((grow & 1) == 0) && (gcol == grow + 1)) {
                        spair[grow >> 1] = s;
                    }
                }
            }
        }
    }

#pragma unroll
    for (int r = 0; r < 8; ++r) {
#pragma unroll
        for (int m = 1; m < 16; m <<= 1)
            rowsum[r] += __shfl_xor(rowsum[r], m, 32);
    }
    if (m16 == 0) {
#pragma unroll
        for (int r = 0; r < 8; ++r)
            atomicAdd(&d[grow_base + 8 * hi16 + r], rowsum[r]);
    }
}

// ---------------- kernel 3: final loss reduction ----------------------------
__global__ __launch_bounds__(256) void cl_reduce_kernel(const float* __restrict__ d,
                                                        const float* __restrict__ spair,
                                                        float* __restrict__ out) {
    __shared__ float sm[256];
    const int tid = threadIdx.x;
    float acc = 0.0f;
    for (int i = tid; i < N_ROWS; i += 256)
        acc += __logf(d[i]) - spair[i >> 1] * TAU_INV;
    sm[tid] = acc;
    __syncthreads();
#pragma unroll
    for (int s = 128; s > 0; s >>= 1) {
        if (tid < s) sm[tid] += sm[tid + s];
        __syncthreads();
    }
    if (tid == 0) out[0] = sm[0] / (float)N_ROWS;
}

extern "C" void kernel_launch(void* const* d_in, const int* in_sizes, int n_in,
                              void* d_out, int out_size, void* d_ws, size_t ws_size,
                              hipStream_t stream) {
    const float* x = (const float*)d_in[0];
    float* out = (float*)d_out;

    // workspace layout: d[8192] f32 | spair[4096] f32 | invn[8192] f32 |
    //                   xhi[8192*256] bf16 | xlo[8192*256] bf16
    float*  d     = (float*)d_ws;
    float*  spair = d + N_ROWS;
    float*  invn  = spair + N_ROWS / 2;
    __bf16* xhi   = (__bf16*)(invn + N_ROWS);
    __bf16* xlo   = xhi + (size_t)N_ROWS * DIM;

    const size_t need = (size_t)(N_ROWS + N_ROWS / 2 + N_ROWS) * sizeof(float)
                      + (size_t)2 * N_ROWS * DIM * sizeof(__bf16);

    cl_zero_kernel<<<(N_ROWS + 255) / 256, 256, 0, stream>>>(d);
    cl_norm_kernel<<<N_ROWS / 8, 256, 0, stream>>>(x, invn);

    dim3 grid(N_ROWS / BM, CSPLIT);
    if (ws_size >= need) {
        // async double-buffered path
        cl_convert_kernel<<<(N_ROWS * DIM / 4) / 256, 256, 0, stream>>>(x, xhi, xlo);
        static bool attr_set = false;
        if (!attr_set) {
            (void)hipFuncSetAttribute((const void*)cl_main_async,
                                      hipFuncAttributeMaxDynamicSharedMemorySize,
                                      2 * (int)BUF_BYTES);
            attr_set = true;
        }
        cl_main_async<<<grid, 256, 2 * BUF_BYTES, stream>>>(xhi, xlo, invn, d, spair);
    } else {
        cl_main_sync<<<grid, 256, BUF_BYTES, stream>>>(x, invn, d, spair);
    }

    cl_reduce_kernel<<<1, 256, 0, stream>>>(d, spair, out);
}
